// SnowCLIPLoss_32916629356881
// MI455X (gfx1250) — compile-verified, matrix operand order checked by reading, not run on hardware
//
#include <hip/hip_runtime.h>
#include <hip/hip_bf16.h>
#include <math.h>

typedef __attribute__((ext_vector_type(16))) __bf16 v16bf;
typedef __attribute__((ext_vector_type(8)))  __bf16 v8bf;
typedef __attribute__((ext_vector_type(8)))  float  v8f;

#define DDIM   768
#define BDIM   256
#define SDIM   65536
#define QSZ    512
#define KSTEPS (DDIM / 32)
#define INV_TEMP 10.0f
#define DEG2RAD 0.017453292519943295f
#define EARTH_R 6371.0088f
#define CUTOFF  25.0f

// ---------- helpers ----------
__device__ __forceinline__ unsigned int fkey(float f) {
  unsigned int u = __float_as_uint(f);
  return (u & 0x80000000u) ? ~u : (u | 0x80000000u);
}
__device__ __forceinline__ unsigned long long shflxor64(unsigned long long x, int m) {
  unsigned int lo = (unsigned int)x;
  unsigned int hi = (unsigned int)(x >> 32);
  lo = __shfl_xor(lo, m, 32);
  hi = __shfl_xor(hi, m, 32);
  return ((unsigned long long)hi << 32) | (unsigned long long)lo;
}
// Build a 16x32 bf16 A/B fragment from two 16B chunks (ISA 16-bit layout:
// lane l holds K = khalf*8 + {0..7} and K = 16 + khalf*8 + {0..7}).
__device__ __forceinline__ v16bf make_frag(const __bf16* p) {
  v8bf lo = *(const v8bf*)(p);
  v8bf hi = *(const v8bf*)(p + 16);
  v16bf r;
#pragma unroll
  for (int i = 0; i < 8; ++i) { r[i] = lo[i]; r[i + 8] = hi[i]; }
  return r;
}

// ---------- kernel 0: init small accumulators ----------
__global__ void k_init(unsigned long long* amax, float* numv, float* bden, float* qden) {
  int t = threadIdx.x;             // 256 threads
  amax[t] = 0ull;
  numv[t] = 0.f;
  qden[t] = 0.f;
#pragma unroll
  for (int j = 0; j < 3; ++j) bden[j * BDIM + t] = 0.f;
}

// ---------- kernel 1: org/augs = normalize(V[src]*L) -> bf16 ----------
__global__ __launch_bounds__(256) void k_prep(const float* __restrict__ V,
                                              const float* __restrict__ L,
                                              __bf16* __restrict__ orgb,
                                              __bf16* __restrict__ augb) {
  __shared__ float red[8];
  int r = blockIdx.x;              // 0..1023
  int src = r >> 8, b = r & 255;
  const float* v = V + ((size_t)src * BDIM + b) * DDIM;
  const float* l = L + (size_t)b * DDIM;
  int tid = threadIdx.x, lane = tid & 31, wave = tid >> 5;
  float x[3]; float ss = 0.f;
#pragma unroll
  for (int i = 0; i < 3; ++i) {
    int d = tid + i * 256;
    x[i] = v[d] * l[d];
    ss += x[i] * x[i];
  }
#pragma unroll
  for (int m = 1; m < 32; m <<= 1) ss += __shfl_xor(ss, m, 32);
  if (lane == 0) red[wave] = ss;
  __syncthreads();
  float tot = 0.f;
#pragma unroll
  for (int w = 0; w < 8; ++w) tot += red[w];
  float scale = 1.f / fmaxf(sqrtf(tot), 1e-12f);
  __bf16* dst = (src == 0) ? (orgb + (size_t)b * DDIM)
                           : (augb + ((size_t)(src - 1) * BDIM + b) * DDIM);
#pragma unroll
  for (int i = 0; i < 3; ++i) dst[tid + i * 256] = (__bf16)(x[i] * scale);
}

// ---------- kernel 2: sim GEMM (bf16 WMMA) + fused argmax ----------
// grid = (SDIM/128, 2); block = 256 (8 waves). Wave w owns b-rows
// [bhalf*128 + w*16, +16) vs this block's 128 support rows.
__global__ __launch_bounds__(256) void k_sim_argmax(const float* __restrict__ sf,
                                                    const __bf16* __restrict__ orgb,
                                                    unsigned long long* __restrict__ amax) {
  __shared__ __attribute__((aligned(16))) __bf16 slab[128 * 32];   // 8 KB per K-step
  const int tid  = threadIdx.x;
  const int lane = tid & 31, wave = tid >> 5;
  const int s0   = blockIdx.x * 128;
  const int bRow0 = blockIdx.y * 128 + wave * 16;
  const int khalf = lane >> 4;
  const int l15   = lane & 15;

  v8f acc[8];
#pragma unroll
  for (int t = 0; t < 8; ++t)
#pragma unroll
    for (int i = 0; i < 8; ++i) acc[t][i] = 0.f;

  for (int ks = 0; ks < KSTEPS; ++ks) {
    const int k0 = ks * 32;
    // stage 128 support rows x 32 K, f32 -> bf16
    {
      int sl = tid >> 1;
      int kl = (tid & 1) * 16;
      const float* src = sf + (size_t)(s0 + sl) * DDIM + k0 + kl;
      __bf16* dst = slab + sl * 32 + kl;
#pragma unroll
      for (int i = 0; i < 16; i += 2) {
        float2 f2 = *(const float2*)(src + i);
        dst[i]     = (__bf16)f2.x;
        dst[i + 1] = (__bf16)f2.y;
      }
    }
    __syncthreads();
    // A fragment straight from global (org is L2-hot)
    v16bf afrag = make_frag(orgb + (size_t)(bRow0 + l15) * DDIM + k0 + khalf * 8);
    // Preload ALL 8 B fragments into distinct registers first so the DS
    // clause issues up-front and WMMAs overlap LDS latency (partial
    // s_wait_dscnt instead of drain-to-zero per tile).
    v16bf bfrag[8];
#pragma unroll
    for (int t = 0; t < 8; ++t)
      bfrag[t] = make_frag(slab + (t * 16 + l15) * 32 + khalf * 8);
#pragma unroll
    for (int t = 0; t < 8; ++t)
      acc[t] = __builtin_amdgcn_wmma_f32_16x16x32_bf16(
          false, afrag, false, bfrag[t], (short)0, acc[t], false, false);
    __syncthreads();
  }

  // fused argmax epilogue: VGPR v holds row bRow0 + v + 8*khalf, col s0+t*16+l15
#pragma unroll
  for (int v = 0; v < 8; ++v) {
    unsigned long long best = 0ull;
#pragma unroll
    for (int t = 0; t < 8; ++t) {
      unsigned int sidx = (unsigned int)(s0 + t * 16 + l15);
      unsigned long long p = ((unsigned long long)fkey(acc[t][v]) << 32) |
                             (unsigned long long)(0xFFFFFFFFu - sidx); // ties -> lowest index
      best = (p > best) ? p : best;
    }
#pragma unroll
    for (int m = 1; m < 16; m <<= 1) {
      unsigned long long o = shflxor64(best, m);
      best = (o > best) ? o : best;
    }
    if (l15 == 0) {
      int row = bRow0 + v + 8 * khalf;
      atomicMax(&amax[row], best);
    }
  }
}

// ---------- kernel 3: nn = normalize(sf[argmax]) -> f32 + bf16 ----------
__global__ __launch_bounds__(256) void k_nn(const float* __restrict__ sf,
                                            const unsigned long long* __restrict__ amax,
                                            float* __restrict__ nnf,
                                            __bf16* __restrict__ nnb) {
  __shared__ float red[8];
  int b = blockIdx.x;
  unsigned int idx = 0xFFFFFFFFu - (unsigned int)(amax[b] & 0xFFFFFFFFull);
  const float* row = sf + (size_t)idx * DDIM;
  int tid = threadIdx.x, lane = tid & 31, wave = tid >> 5;
  float x[3]; float ss = 0.f;
#pragma unroll
  for (int i = 0; i < 3; ++i) { x[i] = row[tid + i * 256]; ss += x[i] * x[i]; }
#pragma unroll
  for (int m = 1; m < 32; m <<= 1) ss += __shfl_xor(ss, m, 32);
  if (lane == 0) red[wave] = ss;
  __syncthreads();
  float tot = 0.f;
#pragma unroll
  for (int w = 0; w < 8; ++w) tot += red[w];
  float scale = 1.f / fmaxf(sqrtf(tot), 1e-12f);
#pragma unroll
  for (int i = 0; i < 3; ++i) {
    int d = tid + i * 256;
    float y = x[i] * scale;
    nnf[(size_t)b * DDIM + d] = y;
    nnb[(size_t)b * DDIM + d] = (__bf16)y;
  }
}

// ---------- kernel 4: batch_den + num via WMMA (nn x augs_j^T) ----------
// grid = (16, 16, 3); block = 1 wave (32 threads), one 16x16 tile each.
__global__ __launch_bounds__(32) void k_batch(const __bf16* __restrict__ nnb,
                                              const __bf16* __restrict__ augb,
                                              float* __restrict__ bden,
                                              float* __restrict__ numv) {
  const int lane  = threadIdx.x;
  const int mt = blockIdx.x, nt = blockIdx.y, j = blockIdx.z;
  const int khalf = lane >> 4, l15 = lane & 15;
  const __bf16* bmat = augb + (size_t)j * BDIM * DDIM;

  v8f acc;
#pragma unroll
  for (int i = 0; i < 8; ++i) acc[i] = 0.f;
  // 2-deep unroll so global fragment loads of step ks+1 overlap WMMA of ks
#pragma unroll 2
  for (int ks = 0; ks < KSTEPS; ++ks) {
    int k0 = ks * 32;
    v16bf a  = make_frag(nnb  + (size_t)(mt * 16 + l15) * DDIM + k0 + khalf * 8);
    v16bf bb = make_frag(bmat + (size_t)(nt * 16 + l15) * DDIM + k0 + khalf * 8);
    acc = __builtin_amdgcn_wmma_f32_16x16x32_bf16(
        false, a, false, bb, (short)0, acc, false, false);
  }
#pragma unroll
  for (int v = 0; v < 8; ++v) {
    int row = mt * 16 + v + 8 * khalf;
    float e = expf(acc[v] * INV_TEMP);
#pragma unroll
    for (int m = 1; m < 16; m <<= 1) e += __shfl_xor(e, m, 32);
    if (l15 == 0) atomicAdd(&bden[j * BDIM + row], e);
    // diagonal -> num (only j==0, matching reference augs[0] = V[1])
    if (j == 0 && mt == nt && l15 == (v + 8 * khalf))
      numv[mt * 16 + l15] = acc[v] * INV_TEMP;
  }
}

// ---------- kernel 5: GPS queue (stable compaction) + queue_den ----------
__global__ __launch_bounds__(256) void k_queue(const float* __restrict__ sf,
                                               const float* __restrict__ sgps,
                                               const float* __restrict__ gps,
                                               const float* __restrict__ nnf,
                                               float* __restrict__ qden) {
  __shared__ int qidx[QSZ];
  __shared__ int woff[8];
  __shared__ int scount;
  int b = blockIdx.x;
  int tid = threadIdx.x, lane = tid & 31, wave = tid >> 5;
  float lat1 = gps[b * 2] * DEG2RAD;
  float lon1 = gps[b * 2 + 1] * DEG2RAD;
  float clat1 = cosf(lat1);
  if (tid == 0) scount = 0;
  __syncthreads();

  for (int sb = 0; sb < SDIM; sb += 256) {
    int s = sb + tid;
    float lat2 = sgps[s * 2] * DEG2RAD;
    float lon2 = sgps[s * 2 + 1] * DEG2RAD;
    float sdl = sinf((lat2 - lat1) * 0.5f);
    float sdo = sinf((lon2 - lon1) * 0.5f);
    float a = sdl * sdl + clat1 * cosf(lat2) * sdo * sdo;
    a = fminf(fmaxf(a, 0.f), 1.f);
    float dist = 2.f * EARTH_R * asinf(sqrtf(a));
    bool m = dist > CUTOFF;
    unsigned long long bal = __ballot(m);
    if (lane == 0) woff[wave] = __popcll(bal);
    __syncthreads();
    int base = scount;
    int wbase = 0;
    for (int w = 0; w < wave; ++w) wbase += woff[w];
    int rank = base + wbase + __popcll(bal & ((1ull << lane) - 1ull));
    if (m && rank < QSZ) qidx[rank] = s;
    __syncthreads();
    if (tid == 0) {
      int tot = 0;
      for (int w = 0; w < 8; ++w) tot += woff[w];
      scount = base + tot;
    }
    __syncthreads();
    if (scount >= QSZ) break;
  }
  int vc = min(scount, QSZ);

  // one wave per queue entry: 768-dim dot, exp, accumulate
  const float* nrow = nnf + (size_t)b * DDIM;
  float wsum = 0.f;
  for (int q = wave; q < vc; q += 8) {
    const float* srow = sf + (size_t)qidx[q] * DDIM;
    float d = 0.f;
    for (int kk = lane; kk < DDIM; kk += 32) d += nrow[kk] * srow[kk];
#pragma unroll
    for (int m = 1; m < 32; m <<= 1) d += __shfl_xor(d, m, 32);
    if (lane == 0) wsum += expf(d * INV_TEMP);
  }
  if (lane == 0) atomicAdd(&qden[b], wsum);
  if (tid == 0) atomicAdd(&qden[b], (float)(QSZ - vc)); // zero-pad rows: exp(0)=1
}

// ---------- kernel 6: final loss reduction ----------
__global__ __launch_bounds__(256) void k_loss(const float* __restrict__ numv,
                                              const float* __restrict__ bden,
                                              const float* __restrict__ qden,
                                              float* __restrict__ out) {
  __shared__ float red[8];
  int b = threadIdx.x, lane = b & 31, wave = b >> 5;
  float a = 0.f;
#pragma unroll
  for (int j = 0; j < 3; ++j)
    a += numv[b] - logf(bden[j * BDIM + b] + qden[b]);
#pragma unroll
  for (int m = 1; m < 32; m <<= 1) a += __shfl_xor(a, m, 32);
  if (lane == 0) red[wave] = a;
  __syncthreads();
  if (b == 0) {
    float tot = 0.f;
    for (int w = 0; w < 8; ++w) tot += red[w];
    out[0] = -tot / (float)BDIM;
  }
}

// ---------- launch ----------
extern "C" void kernel_launch(void* const* d_in, const int* in_sizes, int n_in,
                              void* d_out, int out_size, void* d_ws, size_t ws_size,
                              hipStream_t stream) {
  (void)in_sizes; (void)n_in; (void)out_size; (void)ws_size;
  const float* V    = (const float*)d_in[0];  // (4,256,768)
  const float* L    = (const float*)d_in[1];  // (256,768)
  const float* gps  = (const float*)d_in[2];  // (256,2)
  const float* sf   = (const float*)d_in[3];  // (65536,768)
  const float* sgps = (const float*)d_in[4];  // (65536,2)
  float* out = (float*)d_out;

  char* ws = (char*)d_ws;
  __bf16* orgb = (__bf16*)(ws);                               // 393,216 B
  __bf16* augb = (__bf16*)(ws + 393216);                      // 1,179,648 B
  __bf16* nnb  = (__bf16*)(ws + 1572864);                     // 393,216 B
  float*  nnf  = (float*)(ws + 1966080);                      // 786,432 B
  unsigned long long* amax = (unsigned long long*)(ws + 2752512); // 2,048 B
  float* numv = (float*)(ws + 2754560);                       // 1,024 B
  float* bden = (float*)(ws + 2755584);                       // 3,072 B
  float* qden = (float*)(ws + 2758656);                       // 1,024 B

  k_init<<<1, 256, 0, stream>>>(amax, numv, bden, qden);
  k_prep<<<1024, 256, 0, stream>>>(V, L, orgb, augb);
  k_sim_argmax<<<dim3(SDIM / 128, 2), 256, 0, stream>>>(sf, orgb, amax);
  k_nn<<<BDIM, 256, 0, stream>>>(sf, amax, nnf, nnb);
  k_batch<<<dim3(16, 16, 3), 32, 0, stream>>>(nnb, augb, bden, numv);
  k_queue<<<BDIM, 256, 0, stream>>>(sf, sgps, gps, nnf, qden);
  k_loss<<<1, 256, 0, stream>>>(numv, bden, qden, out);
}
